// Dipole_88270167868087
// MI455X (gfx1250) — matrix operand order repaired and checked by reading, not compiled, not run on hardware
//
#include <hip/hip_runtime.h>
#include <math.h>

typedef __attribute__((ext_vector_type(16))) __bf16        v16bf;
typedef __attribute__((ext_vector_type(8)))  float         v8f;
typedef __attribute__((ext_vector_type(4)))  unsigned int  v4u;
typedef __attribute__((ext_vector_type(4)))  float         v4f;
typedef __attribute__((ext_vector_type(2)))  unsigned int  v2u;

union FragAB { v16bf bf; v4u q[2]; };

__device__ __forceinline__ unsigned short f2bf(float f) {
    unsigned u = __builtin_bit_cast(unsigned, f);
    u += 0x7FFFu + ((u >> 16) & 1u);          // RNE
    return (unsigned short)(u >> 16);
}
__device__ __forceinline__ float bf2f(unsigned short h) {
    unsigned u = ((unsigned)h) << 16;
    return __builtin_bit_cast(float, u);
}
__device__ __forceinline__ float silu(float x) { return x / (1.0f + __expf(-x)); }

// pack 4 floats -> 4 bf16 in 2 dwords
__device__ __forceinline__ v2u pack4bf(v4f v) {
    v2u r;
    r.x = (unsigned)f2bf(v.x) | ((unsigned)f2bf(v.y) << 16);
    r.y = (unsigned)f2bf(v.z) | ((unsigned)f2bf(v.w) << 16);
    return r;
}

// A-fragment (16x32 bf16) from LDS, row-major [row][K]:
// lane l: M = l&15 ; halves 0..7 -> K = kbase + 8*(l>>4) + i ; halves 8..15 -> +16
__device__ __forceinline__ void loadA(FragAB& f, const unsigned short* sA,
                                      int rowbase, int strideA, int kbase, int lane) {
    const unsigned short* p = sA + (rowbase + (lane & 15)) * strideA + kbase + 8 * (lane >> 4);
    f.q[0] = *(const v4u*)(p);
    f.q[1] = *(const v4u*)(p + 16);
}
// B-fragment (32x16 bf16) from global, weights stored transposed [N][K]:
// lane l: N = colbase + (l&15) ; halves 0..7 -> K = kbase + 8*(l>>4) + i ; halves 8..15 -> +16
__device__ __forceinline__ void loadB(FragAB& f, const unsigned short* wT,
                                      int colbase, int strideB, int kbase, int lane) {
    const unsigned short* p = wT + (size_t)(colbase + (lane & 15)) * strideB + kbase + 8 * (lane >> 4);
    f.q[0] = *(const v4u*)(p);
    f.q[1] = *(const v4u*)(p + 16);
}

template<int KSTEPS>
__device__ __forceinline__ v8f mm_tile(const unsigned short* sA, int rowbase, int strideA,
                                       const unsigned short* wT, int colbase, int strideB,
                                       int lane) {
    v8f acc = {0.f,0.f,0.f,0.f,0.f,0.f,0.f,0.f};
#pragma unroll
    for (int ks = 0; ks < KSTEPS; ++ks) {
        FragAB fa, fb;
        loadA(fa, sA, rowbase, strideA, ks * 32, lane);
        loadB(fb, wT, colbase, strideB, ks * 32, lane);
        acc = __builtin_amdgcn_wmma_f32_16x16x32_bf16(false, fa.bf, false, fb.bf,
                                                      (short)0, acc, false, false);
    }
    return acc;
}

// ---- workspace layout (ushort units) ----
// Wmix1T [512][256] @ 0 ; W1aT [256][512] @ 131072 ; W2aT [512][256] @ 262144 ;
// W1bT [256][512] (K<257 real, rest 0) @ 393216.  Total 1 MB.
#define WM1T 0
#define W1AT 131072
#define W2AT 262144
#define W1BT 393216

__global__ void prep_weights(const float* __restrict__ Wmix1, const float* __restrict__ W1a,
                             const float* __restrict__ W2a,   const float* __restrict__ W1b,
                             unsigned short* __restrict__ ws) {
    int idx = blockIdx.x * 256 + threadIdx.x;
    if (idx < 512 * 256) { int n = idx / 256, k = idx % 256;
        ws[WM1T + idx] = f2bf(Wmix1[k * 512 + n]); return; }
    idx -= 512 * 256;
    if (idx < 256 * 512) { int n = idx / 512, k = idx % 512;
        ws[W1AT + idx] = f2bf(W1a[k * 256 + n]); return; }
    idx -= 256 * 512;
    if (idx < 512 * 256) { int n = idx / 256, k = idx % 256;
        ws[W2AT + idx] = f2bf(W2a[k * 512 + n]); return; }
    idx -= 512 * 256;
    if (idx < 256 * 512) { int n = idx / 512, k = idx % 512;
        ws[W1BT + idx] = (k < 257) ? f2bf(W1b[k * 256 + n]) : (unsigned short)0; return; }
}

__global__ void zero_out(float* __restrict__ out, int n) {
    int i = blockIdx.x * 256 + threadIdx.x;
    if (i < n) out[i] = 0.f;
}

// Fused PaiNN gated blocks: 32 atoms / workgroup, 8 waves.
__global__ __launch_bounds__(256)
void painn_fused(const float* __restrict__ pos, const float* __restrict__ l0,
                 const float* __restrict__ l1,  const int* __restrict__ batch,
                 const unsigned short* __restrict__ ws,
                 const float* __restrict__ b1a, const float* __restrict__ b2a,
                 const float* __restrict__ Wmix2,
                 const float* __restrict__ b1b, const float* __restrict__ W2b,
                 const float* __restrict__ b2b,
                 float* __restrict__ out, int natoms, int nmol) {
    __shared__ __attribute__((aligned(16))) unsigned char smem[136192];
    unsigned short* sL1  = (unsigned short*)smem;            // 96x256 bf16 (stage-1 A)
    unsigned short* sH   = (unsigned short*)smem;            // alias: 32x256 bf16 (h)
    float*          sGATE= (float*)(smem + 16384);           // alias: 32x256 f32 (gate)
    unsigned short* sCTX = (unsigned short*)(smem + 49152);  // 32x512 bf16 (ctx / ctx2)
    unsigned short* sVW  = (unsigned short*)(smem + 81920);  // 96x256 bf16 (vW)
    float* sV2  = (float*)(smem + 131072);                   // 96: vV2
    float* sW2v = sV2 + 96;                                  // 96: vW2
    float* sCHG = sW2v + 96;                                 // 32: charge
    float* sGT2 = sCHG + 32;                                 // 32: gate2
    float* sWM2 = (float*)(smem + 132096);                   // 512: Wmix2 cache
    float* sW2B = (float*)(smem + 134144);                   // 512: W2b  cache

    const int tid  = threadIdx.x;
    const int lane = tid & 31;
    const int w    = tid >> 5;
    const int a0   = blockIdx.x * 32;
    const int rl   = (lane >> 4) * 8;
    const int col  = lane & 15;

    // ---- Stage 0: stage l1 (rows ordered c*32+n) and l0 -> bf16 LDS ----
    if (a0 + 32 <= natoms) {
        // fast path: tile fully contiguous in global -> float4 loads, b64 LDS stores
        const v4f* l1v = (const v4f*)(l1 + (size_t)a0 * 768);
        for (int e4 = tid; e4 < 6144; e4 += 256) {
            int g = e4 * 4;
            int n = g / 768, rem = g - n * 768, c = rem >> 8, i = rem & 255;
            v4f v = l1v[e4];
            *(v2u*)(sL1 + (c * 32 + n) * 256 + i) = pack4bf(v);
        }
        const v4f* l0v = (const v4f*)(l0 + (size_t)a0 * 256);
        for (int e4 = tid; e4 < 2048; e4 += 256) {
            int g = e4 * 4, n = g >> 8, i = g & 255;
            *(v2u*)(sCTX + n * 512 + i) = pack4bf(l0v[e4]);
        }
    } else {
        for (int e = tid; e < 32 * 3 * 256; e += 256) {
            int n = e / 768, rem = e - n * 768, c = rem >> 8, i = rem & 255;
            int a = a0 + n; if (a >= natoms) a = natoms - 1;
            sL1[(c * 32 + n) * 256 + i] = f2bf(l1[((size_t)a * 3 + c) * 256 + i]);
        }
        for (int e = tid; e < 32 * 256; e += 256) {
            int n = e >> 8, i = e & 255;
            int a = a0 + n; if (a >= natoms) a = natoms - 1;
            sCTX[n * 512 + i] = f2bf(l0[(size_t)a * 256 + i]);
        }
    }
    // cache the tiny fp32 weights used by VALU stages
    sWM2[tid] = Wmix2[tid]; sWM2[tid + 256] = Wmix2[tid + 256];
    sW2B[tid] = W2b[tid];   sW2B[tid + 256] = W2b[tid + 256];
    __syncthreads();

    // ---- Stage 1: vmix = v @ Wmix1 ; vVn -> ctx[:,256:512] ; vW -> sVW ----
    for (int j = 0; j < 8; ++j) {
        int item = j * 8 + w;                 // 64 items = 2 n-tiles x 32 o-tiles
        int nt = item & 1, ot = item >> 1;
        v8f c0 = {0,0,0,0,0,0,0,0}, c1 = c0, c2 = c0;
#pragma unroll
        for (int ks = 0; ks < 8; ++ks) {
            FragAB fb, fa;
            loadB(fb, ws + WM1T, ot * 16, 256, ks * 32, lane);
            loadA(fa, sL1,      nt * 16, 256, ks * 32, lane);
            c0 = __builtin_amdgcn_wmma_f32_16x16x32_bf16(false, fa.bf, false, fb.bf, (short)0, c0, false, false);
            loadA(fa, sL1, 32 + nt * 16, 256, ks * 32, lane);
            c1 = __builtin_amdgcn_wmma_f32_16x16x32_bf16(false, fa.bf, false, fb.bf, (short)0, c1, false, false);
            loadA(fa, sL1, 64 + nt * 16, 256, ks * 32, lane);
            c2 = __builtin_amdgcn_wmma_f32_16x16x32_bf16(false, fa.bf, false, fb.bf, (short)0, c2, false, false);
        }
        if (ot < 16) {                        // vV half -> norm over the 3 rows
#pragma unroll
            for (int r = 0; r < 8; ++r) {
                int n = nt * 16 + r + rl;
                float s = c0[r]*c0[r] + c1[r]*c1[r] + c2[r]*c2[r];
                sCTX[n * 512 + 256 + ot * 16 + col] = f2bf(sqrtf(s));
            }
        } else {                              // vW half -> keep (bf16)
#pragma unroll
            for (int r = 0; r < 8; ++r) {
                int n = nt * 16 + r + rl, oc = (ot - 16) * 16 + col;
                sVW[(n)      * 256 + oc] = f2bf(c0[r]);
                sVW[(32 + n) * 256 + oc] = f2bf(c1[r]);
                sVW[(64 + n) * 256 + oc] = f2bf(c2[r]);
            }
        }
    }
    __syncthreads();

    // ---- Stage 2: h = silu(ctx @ W1a + b1a) ----
    for (int j = 0; j < 4; ++j) {
        int item = j * 8 + w, nt = item & 1, ot = item >> 1;  // 2 x 16 tiles
        v8f acc = mm_tile<16>(sCTX, nt * 16, 512, ws + W1AT, ot * 16, 512, lane);
#pragma unroll
        for (int r = 0; r < 8; ++r) {
            int n = nt * 16 + r + rl, oc = ot * 16 + col;
            sH[n * 256 + oc] = f2bf(silu(acc[r] + b1a[oc]));
        }
    }
    __syncthreads();

    // ---- Stage 3: x = h @ W2a + b2a ; s2=silu(x[:,:256]) ; gate=x[:,256:] ----
    for (int j = 0; j < 8; ++j) {
        int item = j * 8 + w, nt = item & 1, ot = item >> 1;  // 2 x 32 tiles
        v8f acc = mm_tile<8>(sH, nt * 16, 256, ws + W2AT, ot * 16, 256, lane);
#pragma unroll
        for (int r = 0; r < 8; ++r) {
            int n = nt * 16 + r + rl, oc = ot * 16 + col;
            float v = acc[r] + b2a[oc];
            if (ot < 16) sCTX[n * 512 + oc]          = f2bf(silu(v));
            else         sGATE[n * 256 + (oc - 256)] = v;
        }
    }
    __syncthreads();

    // ---- Stage 4: vmix2 = (gate*vW) @ Wmix2 ; vVn2 -> ctx2[:,256], pad to 288 ----
    if (tid < 192) {
        int row = tid >> 1, cx = tid & 1, n = row & 31;
        int rot = (row & 63) * 4;             // bank-conflict stagger
        float acc = 0.f;
        for (int ii = 0; ii < 256; ++ii) {
            int i = (ii + rot) & 255;
            acc += sGATE[n * 256 + i] * bf2f(sVW[row * 256 + i]) * sWM2[i * 2 + cx];
        }
        if (cx == 0) sV2[row] = acc; else sW2v[row] = acc;
    }
    __syncthreads();
    if (tid < 32) {
        int n = tid;
        float s = sV2[n]*sV2[n] + sV2[32+n]*sV2[32+n] + sV2[64+n]*sV2[64+n];
        sCTX[n * 512 + 256] = f2bf(sqrtf(s));
        for (int k = 257; k < 288; ++k) sCTX[n * 512 + k] = 0;
    }
    __syncthreads();

    // ---- Stage 5: h2 = silu(ctx2 @ W1b + b1b), K padded 257->288 ----
    for (int j = 0; j < 4; ++j) {
        int item = j * 8 + w, nt = item & 1, ot = item >> 1;
        v8f acc = mm_tile<9>(sCTX, nt * 16, 512, ws + W1BT, ot * 16, 512, lane);
#pragma unroll
        for (int r = 0; r < 8; ++r) {
            int n = nt * 16 + r + rl, oc = ot * 16 + col;
            sH[n * 256 + oc] = f2bf(silu(acc[r] + b1b[oc]));
        }
    }
    __syncthreads();

    // ---- Stage 6: x2 = h2 @ W2b + b2b -> charge, gate2 ----
    if (tid < 64) {
        int n = tid >> 1, cx = tid & 1;
        int rot = (n & 63) * 4;
        float acc = b2b[cx];
        for (int ii = 0; ii < 256; ++ii) {
            int i = (ii + rot) & 255;
            acc += bf2f(sH[n * 256 + i]) * sW2B[i * 2 + cx];
        }
        if (cx == 0) sCHG[n] = acc; else sGT2[n] = acc;
    }
    __syncthreads();

    // ---- Stage 7: v_final = gate2*vW2 ; y = v_final + pos*charge ; segment-sum ----
    if (tid < 96) {
        int row = tid, c = row >> 5, n = row & 31;
        int a = a0 + n;
        if (a < natoms) {
            float vf = sGT2[n] * sW2v[row];
            float yv = vf + pos[(size_t)a * 3 + c] * sCHG[n];
            int mol = batch[a];
            atomicAdd(out + (size_t)mol * 3 + c, yv);
            atomicAdd(out + (size_t)nmol * 3 + (size_t)mol * 3 + c, vf);
        }
    }
}

extern "C" void kernel_launch(void* const* d_in, const int* in_sizes, int n_in,
                              void* d_out, int out_size, void* d_ws, size_t ws_size,
                              hipStream_t stream) {
    const float* pos   = (const float*)d_in[0];
    const float* l0    = (const float*)d_in[1];
    const float* l1    = (const float*)d_in[2];
    const int*   batch = (const int*)d_in[3];
    const float* Wmix1 = (const float*)d_in[4];
    const float* W1a   = (const float*)d_in[5];
    const float* b1a   = (const float*)d_in[6];
    const float* W2a   = (const float*)d_in[7];
    const float* b2a   = (const float*)d_in[8];
    const float* Wmix2 = (const float*)d_in[9];
    const float* W1b   = (const float*)d_in[10];
    const float* b1b   = (const float*)d_in[11];
    const float* W2b   = (const float*)d_in[12];
    const float* b2b   = (const float*)d_in[13];
    (void)n_in; (void)ws_size;

    int natoms = in_sizes[0] / 3;
    int nmol   = out_size / 6;
    unsigned short* ws = (unsigned short*)d_ws;
    float* out = (float*)d_out;

    zero_out<<<(out_size + 255) / 256, 256, 0, stream>>>(out, out_size);
    prep_weights<<<2048, 256, 0, stream>>>(Wmix1, W1a, W2a, W1b, ws);
    int ngrid = (natoms + 31) / 32;
    painn_fused<<<ngrid, 256, 0, stream>>>(pos, l0, l1, batch, ws,
                                           b1a, b2a, Wmix2, b1b, W2b, b2b,
                                           out, natoms, nmol);
}
